// AttentionBlock_73967926772279
// MI455X (gfx1250) — matrix operand order repaired and checked by reading, not compiled, not run on hardware
//
#include <hip/hip_runtime.h>

#define NGROUPS 32
#define EPSV 1e-5f
#define CC 256
#define NN 4096
#define BB 4

typedef __attribute__((ext_vector_type(16))) _Float16 v16h;
typedef __attribute__((ext_vector_type(8)))  _Float16 v8h;
typedef __attribute__((ext_vector_type(8)))  float    v8f;
typedef __attribute__((ext_vector_type(4)))  float    v4f;
typedef __attribute__((ext_vector_type(4)))  unsigned v4u;
typedef __attribute__((ext_vector_type(8)))  int      v8i;
typedef __attribute__((ext_vector_type(4)))  int      v4i;

#define WMMA_F32_F16(a, b, c) \
  __builtin_amdgcn_wmma_f32_16x16x32_f16(false, (a), false, (b), (short)0, (c), false, false)

__device__ __forceinline__ v8f v8f_zero() {
  v8f z;
#pragma unroll
  for (int i = 0; i < 8; ++i) z[i] = 0.f;
  return z;
}

// Load a 16-element f16 fragment from two chunks at p and p+gap (in halfs).
// A-fragment: gap = 8  (contiguous 16 halfs)
// B-fragment: gap = 8  (contiguous 16 halfs)
__device__ __forceinline__ v16h frag2(const _Float16* p, int gap) {
  v8h lo = *(const v8h*)p;
  v8h hi = *(const v8h*)(p + gap);
  v16h r;
#pragma unroll
  for (int i = 0; i < 8; ++i) { r[i] = lo[i]; r[i + 8] = hi[i]; }
  return r;
}

// ---- Tensor Data Mover descriptor helpers (CDNA5 ISA) ----
// Group 0: [1:0]=count=1, [63:32]=lds_addr, [120:64]=global_addr, [127:126]=type=2
__device__ __forceinline__ v4u tdm_g0(unsigned lds_addr, unsigned long long gaddr) {
  v4u g;
  g[0] = 1u;
  g[1] = lds_addr;
  g[2] = (unsigned)(gaddr & 0xFFFFFFFFull);
  g[3] = (unsigned)((gaddr >> 32) & 0x1FFFFFFull) | (2u << 30);
  return g;
}
// Group 1: [17:16]=data_size(1->2B), [79:48]=tensor_dim0, [111:80]=tensor_dim1,
// [127:112]=tile_dim0, [143:128]=tile_dim1, [207:160]=tensor_dim0_stride
__device__ __forceinline__ v8i tdm_g1(unsigned td0, unsigned td1,
                                      unsigned tile0, unsigned tile1, unsigned stride0) {
  v8i g;
  g[0] = (int)(1u << 16);  // data_size = 2 bytes; workgroup_mask = 0
  g[1] = (int)((td0 & 0xFFFFu) << 16);
  g[2] = (int)((td0 >> 16) | ((td1 & 0xFFFFu) << 16));
  g[3] = (int)((td1 >> 16) | ((tile0 & 0xFFFFu) << 16));
  g[4] = (int)(tile1 & 0xFFFFu);
  g[5] = (int)stride0;
  g[6] = 0;
  g[7] = 0;
  return g;
}

__device__ __forceinline__ unsigned lds_off(const void* p) {
  // flat aperture: addr[31:0] is the LDS byte address (ISA 10.2 aperture table)
  return (unsigned)(unsigned long long)(uintptr_t)p;
}

// Issue async TDM loads for one 32-key tile of K (contiguous) and V (strided rows).
__device__ __forceinline__ void tdm_issue_tiles(const _Float16* gk, unsigned ldsk,
                                                const _Float16* gv, unsigned ldsv) {
  v4i z4;
  z4[0] = 0; z4[1] = 0; z4[2] = 0; z4[3] = 0;
  v8i z8;
#pragma unroll
  for (int i = 0; i < 8; ++i) z8[i] = 0;
  // K tile: 32 rows x 256 halfs, contiguous (row stride 256)
  __builtin_amdgcn_tensor_load_to_lds(
      tdm_g0(ldsk, (unsigned long long)(uintptr_t)gk),
      tdm_g1(256u, 32u, 256u, 32u, 256u), z4, z4, z8, 0);
  // V tile: 256 rows x 32 halfs, row stride NN (channel-major source) -> packed LDS
  __builtin_amdgcn_tensor_load_to_lds(
      tdm_g0(ldsv, (unsigned long long)(uintptr_t)gv),
      tdm_g1(32u, 256u, 32u, 256u, (unsigned)NN), z4, z4, z8, 0);
}

// ---------------- K0: convert the 4 weight matrices to f16 ----------------
__global__ void k_cvt_w(const float* qw, const float* kw, const float* vw,
                        const float* pw, _Float16* dst) {
  int idx = blockIdx.x * 256 + threadIdx.x;
  if (idx >= 4 * CC * CC) return;
  int m = idx >> 16, l = idx & 65535;
  const float* s = (m == 0) ? qw : (m == 1) ? kw : (m == 2) ? vw : pw;
  dst[idx] = (_Float16)s[l];
}

// ---------------- K1a: GroupNorm stats -> per-channel scale/shift ----------
__global__ void k_gn_stats(const float* x, const float* gw, const float* gb,
                           float* scale, float* shift) {
  int g = blockIdx.x, b = blockIdx.y, t = threadIdx.x;
  const float* base = x + ((size_t)(b * CC + g * 8)) * NN;
  float s = 0.f, s2 = 0.f;
  for (int i = t; i < 8 * NN; i += 256) { float v = base[i]; s += v; s2 += v * v; }
  __shared__ float r1[256], r2[256];
  r1[t] = s; r2[t] = s2; __syncthreads();
  for (int o = 128; o > 0; o >>= 1) {
    if (t < o) { r1[t] += r1[t + o]; r2[t] += r2[t + o]; }
    __syncthreads();
  }
  if (t < 8) {
    float inv_n = 1.f / (8.f * NN);
    float mu = r1[0] * inv_n;
    float var = r2[0] * inv_n - mu * mu;
    float rstd = rsqrtf(var + EPSV);
    int c = g * 8 + t;
    float sc = gw[c] * rstd;
    scale[b * CC + c] = sc;
    shift[b * CC + c] = gb[c] - mu * sc;
  }
}

// ---------------- K1b: normalize + transpose -> xn[b][n][c] (f16) ----------
__global__ void k_gn_apply(const float* x, const float* scale, const float* shift,
                           _Float16* xn) {
  __shared__ float tile[32][33];
  int n0 = blockIdx.x * 32, c0 = blockIdx.y * 32, b = blockIdx.z;
  int tx = threadIdx.x, ty = threadIdx.y;
#pragma unroll
  for (int i = 0; i < 4; ++i) {
    int c = c0 + ty + i * 8;
    tile[ty + i * 8][tx] = x[((size_t)(b * CC + c)) * NN + n0 + tx];
  }
  __syncthreads();
  int c = c0 + tx;
  float sc = scale[b * CC + c], sh = shift[b * CC + c];
#pragma unroll
  for (int i = 0; i < 4; ++i) {
    int n = n0 + ty + i * 8;
    xn[((size_t)(b * NN + n)) * CC + c] = (_Float16)(tile[tx][ty + i * 8] * sc + sh);
  }
}

// ---------------- K2: Q/K/V projections (WMMA GEMM) ------------------------
// out[o,n] = sum_c W[o,c]*xn[n,c] + bias[o]; q,k -> [N,C], v -> [C,N]
__global__ void __launch_bounds__(256) k_qkv(
    const _Float16* wq, const _Float16* wk, const _Float16* wv,
    const float* qb, const float* kb, const float* vb,
    const _Float16* xn, _Float16* qo, _Float16* ko, _Float16* vo) {
  int n0 = blockIdx.x * 64;
  int which = blockIdx.y;
  int b = blockIdx.z;
  int lane = threadIdx.x & 31, wid = threadIdx.x >> 5;
  int wm = wid & 3, wn = wid >> 2;
  int la = lane & 15, lh = lane >> 4;
  const _Float16* w = (which == 0) ? wq : (which == 1) ? wk : wv;
  const float* bias = (which == 0) ? qb : (which == 1) ? kb : vb;

  v8f acc[4][2];
#pragma unroll
  for (int i = 0; i < 4; ++i)
#pragma unroll
    for (int j = 0; j < 2; ++j) acc[i][j] = v8f_zero();

  const _Float16* xnb = xn + ((size_t)b) * NN * CC;
#pragma unroll
  for (int kc = 0; kc < 8; ++kc) {
    v16h af[4], bf[2];
#pragma unroll
    for (int i = 0; i < 4; ++i)
      af[i] = frag2(w + (size_t)(wm * 64 + 16 * i + la) * CC + kc * 32 + lh * 16, 8);
#pragma unroll
    for (int j = 0; j < 2; ++j)
      bf[j] = frag2(xnb + (size_t)(n0 + wn * 32 + 16 * j + la) * CC + kc * 32 + lh * 16, 8);
#pragma unroll
    for (int i = 0; i < 4; ++i)
#pragma unroll
      for (int j = 0; j < 2; ++j)
        acc[i][j] = WMMA_F32_F16(af[i], bf[j], acc[i][j]);
  }

  float qscale = (which == 0) ? 0.0625f : 1.0f;  // fold C^-0.5 into q
#pragma unroll
  for (int i = 0; i < 4; ++i) {
    int ofirst = wm * 64 + 16 * i + lh * 8;
    v4f b0 = *(const v4f*)(bias + ofirst);
    v4f b1 = *(const v4f*)(bias + ofirst + 4);
#pragma unroll
    for (int j = 0; j < 2; ++j) {
      int n = n0 + wn * 32 + 16 * j + la;
      if (which < 2) {
        v8h outv;
#pragma unroll
        for (int v = 0; v < 8; ++v) {
          float bv = (v < 4) ? b0[v] : b1[v - 4];
          outv[v] = (_Float16)((acc[i][j][v] + bv) * qscale);
        }
        _Float16* dst = ((which == 0) ? qo : ko) + ((size_t)(b * NN + n)) * CC + ofirst;
        *(v8h*)dst = outv;
      } else {
#pragma unroll
        for (int v = 0; v < 8; ++v) {
          float bv = (v < 4) ? b0[v] : b1[v - 4];
          vo[((size_t)(b * CC + ofirst + v)) * NN + n] = (_Float16)(acc[i][j][v] + bv);
        }
      }
    }
  }
}

// ---------------- K3: flash attention, TDM double-buffered K/V tiles --------
__global__ void __launch_bounds__(256) k_attn(const _Float16* q, const _Float16* k,
                                              const _Float16* v, _Float16* hout) {
  __shared__ _Float16 klds[2][32 * 256];   // [buf][key][c]
  __shared__ _Float16 vtlds[2][256 * 32];  // [buf][c][key]
  __shared__ _Float16 plds[8 * 16 * 32];
  int b = blockIdx.y;
  int tid = threadIdx.x;
  int lane = tid & 31, wid = tid >> 5;
  int la = lane & 15, lh = lane >> 4;
  int row_base = blockIdx.x * 128 + wid * 16;
  const int T = NN / 32;

  const _Float16* kbase = k + ((size_t)b) * NN * CC;
  const _Float16* vbase = v + ((size_t)b) * CC * NN;
  unsigned koff[2], voff[2];
  koff[0] = lds_off(&klds[0][0]);  koff[1] = lds_off(&klds[1][0]);
  voff[0] = lds_off(&vtlds[0][0]); voff[1] = lds_off(&vtlds[1][0]);

  v16h qf[8];
  const _Float16* qrow = q + ((size_t)(b * NN + row_base + la)) * CC;
#pragma unroll
  for (int kc = 0; kc < 8; ++kc) qf[kc] = frag2(qrow + kc * 32 + lh * 16, 8);

  v8f acc[16];
#pragma unroll
  for (int t = 0; t < 16; ++t) acc[t] = v8f_zero();
  float m_run[8], l_run[8];
#pragma unroll
  for (int r = 0; r < 8; ++r) { m_run[r] = -3.0e38f; l_run[r] = 0.f; }

  // prologue: async-load tile 0 into buffer 0
  if (wid == 0) {
    tdm_issue_tiles(kbase, koff[0], vbase, voff[0]);
  }

  for (int kt = 0; kt < T; ++kt) {
    int cur = kt & 1;
    if (wid == 0) {
      if (kt + 1 < T) {
        // prefetch next tile into the other buffer, then wait only for the
        // older pair (TENSORcnt retires in order)
        tdm_issue_tiles(kbase + (size_t)(kt + 1) * 32 * CC, koff[cur ^ 1],
                        vbase + (size_t)(kt + 1) * 32, voff[cur ^ 1]);
        __builtin_amdgcn_s_wait_tensorcnt(2);
      } else {
        __builtin_amdgcn_s_wait_tensorcnt(0);
      }
    }
    __syncthreads();  // current tile visible to all waves

    const _Float16* kt_lds = &klds[cur][0];
    const _Float16* vt_lds = &vtlds[cur][0];

    v8f s0 = v8f_zero(), s1 = v8f_zero();
#pragma unroll
    for (int kc = 0; kc < 8; ++kc) {
      v16h bk0 = frag2(kt_lds + la * 256 + kc * 32 + lh * 16, 8);
      v16h bk1 = frag2(kt_lds + (16 + la) * 256 + kc * 32 + lh * 16, 8);
      s0 = WMMA_F32_F16(qf[kc], bk0, s0);
      s1 = WMMA_F32_F16(qf[kc], bk1, s1);
    }

    float esc[8];
#pragma unroll
    for (int r = 0; r < 8; ++r) {
      float mx = fmaxf(s0[r], s1[r]);
#pragma unroll
      for (int off = 1; off < 16; off <<= 1) mx = fmaxf(mx, __shfl_xor(mx, off, 16));
      float nm = fmaxf(m_run[r], mx);
      esc[r] = __expf(m_run[r] - nm);
      m_run[r] = nm;
      float p0 = __expf(s0[r] - nm);
      float p1 = __expf(s1[r] - nm);
      s0[r] = p0; s1[r] = p1;
      float rs = p0 + p1;
#pragma unroll
      for (int off = 1; off < 16; off <<= 1) rs += __shfl_xor(rs, off, 16);
      l_run[r] = l_run[r] * esc[r] + rs;
    }
#pragma unroll
    for (int t = 0; t < 16; ++t)
#pragma unroll
      for (int r = 0; r < 8; ++r) acc[t][r] *= esc[r];

    // D-layout P -> A-layout via per-wave LDS bounce
    _Float16* pw = plds + wid * 512;
#pragma unroll
    for (int r = 0; r < 8; ++r) {
      int row = r + lh * 8;
      pw[row * 32 + la] = (_Float16)s0[r];
      pw[row * 32 + 16 + la] = (_Float16)s1[r];
    }
    v16h pa = frag2(pw + la * 32 + lh * 16, 8);
#pragma unroll
    for (int ct = 0; ct < 16; ++ct) {
      v16h bf = frag2(vt_lds + (ct * 16 + la) * 32 + lh * 16, 8);
      acc[ct] = WMMA_F32_F16(pa, bf, acc[ct]);
    }
    __syncthreads();  // all reads of buf[cur] done before TDM overwrites it
  }

  float inv_l[8];
#pragma unroll
  for (int r = 0; r < 8; ++r) inv_l[r] = 1.f / l_run[r];
#pragma unroll
  for (int ct = 0; ct < 16; ++ct) {
    int c = ct * 16 + la;
#pragma unroll
    for (int r = 0; r < 8; ++r) {
      int row = row_base + r + lh * 8;
      hout[((size_t)(b * NN + row)) * CC + c] = (_Float16)(acc[ct][r] * inv_l[r]);
    }
  }
}

// ---------------- K4: output projection + bias + residual ------------------
// out[b][o][n] = sum_c h[n,c]*pw[o,c] + pb[o] + x[b][o][n]
__global__ void __launch_bounds__(256) k_proj(const _Float16* hin, const _Float16* wp,
                                              const float* pb, const float* x, float* out) {
  int b = blockIdx.y;
  int brow = blockIdx.x * 128;
  int lane = threadIdx.x & 31, wid = threadIdx.x >> 5;
  int la = lane & 15, lh = lane >> 4;
  int wr = wid >> 2, wc = wid & 3;

  v8f acc[4][4];
#pragma unroll
  for (int i = 0; i < 4; ++i)
#pragma unroll
    for (int j = 0; j < 4; ++j) acc[i][j] = v8f_zero();

#pragma unroll
  for (int kc = 0; kc < 8; ++kc) {
    v16h af[4], bf[4];
#pragma unroll
    for (int i = 0; i < 4; ++i)
      af[i] = frag2(hin + ((size_t)(b * NN + brow + wr * 64 + 16 * i + la)) * CC + kc * 32 + lh * 16, 8);
#pragma unroll
    for (int j = 0; j < 4; ++j)
      bf[j] = frag2(wp + (size_t)(wc * 64 + 16 * j + la) * CC + kc * 32 + lh * 16, 8);
#pragma unroll
    for (int i = 0; i < 4; ++i)
#pragma unroll
      for (int j = 0; j < 4; ++j)
        acc[i][j] = WMMA_F32_F16(af[i], bf[j], acc[i][j]);
  }

#pragma unroll
  for (int i = 0; i < 4; ++i)
#pragma unroll
    for (int j = 0; j < 4; ++j) {
      int o = wc * 64 + 16 * j + la;
      int nf = brow + wr * 64 + 16 * i + lh * 8;
      float bias = pb[o];
      const float* xp = x + ((size_t)(b * CC + o)) * NN + nf;
      float* op = out + ((size_t)(b * CC + o)) * NN + nf;
      v4f x0 = *(const v4f*)xp, x1 = *(const v4f*)(xp + 4);
      v4f r0, r1;
#pragma unroll
      for (int vv = 0; vv < 4; ++vv) {
        r0[vv] = acc[i][j][vv] + bias + x0[vv];
        r1[vv] = acc[i][j][vv + 4] + bias + x1[vv];
      }
      *(v4f*)op = r0;
      *(v4f*)(op + 4) = r1;
    }
}

// ---------------------------------------------------------------------------
extern "C" void kernel_launch(void* const* d_in, const int* in_sizes, int n_in,
                              void* d_out, int out_size, void* d_ws, size_t ws_size,
                              hipStream_t stream) {
  const float* x    = (const float*)d_in[0];
  const float* gn_w = (const float*)d_in[1];
  const float* gn_b = (const float*)d_in[2];
  const float* q_w  = (const float*)d_in[3];
  const float* q_b  = (const float*)d_in[4];
  const float* k_w  = (const float*)d_in[5];
  const float* k_b  = (const float*)d_in[6];
  const float* v_w  = (const float*)d_in[7];
  const float* v_b  = (const float*)d_in[8];
  const float* p_w  = (const float*)d_in[9];
  const float* p_b  = (const float*)d_in[10];

  const size_t ACT = (size_t)BB * NN * CC;  // elements per f16 activation buffer
  _Float16* wqh = (_Float16*)d_ws;
  _Float16* wkh = wqh + 65536;
  _Float16* wvh = wkh + 65536;
  _Float16* wph = wvh + 65536;
  _Float16* xn  = wph + 65536;
  _Float16* qws = xn + ACT;
  _Float16* kws = qws + ACT;
  _Float16* vws = kws + ACT;
  _Float16* hws = vws + ACT;
  float* scale = (float*)(hws + ACT);
  float* shift = scale + BB * CC;

  k_cvt_w<<<(4 * CC * CC + 255) / 256, 256, 0, stream>>>(q_w, k_w, v_w, p_w, wqh);
  k_gn_stats<<<dim3(NGROUPS, BB), 256, 0, stream>>>(x, gn_w, gn_b, scale, shift);
  k_gn_apply<<<dim3(NN / 32, CC / 32, BB), dim3(32, 8), 0, stream>>>(x, scale, shift, xn);
  k_qkv<<<dim3(NN / 64, 3, BB), 256, 0, stream>>>(wqh, wkh, wvh, q_b, k_b, v_b,
                                                  xn, qws, kws, vws);
  k_attn<<<dim3(NN / 128, BB), 256, 0, stream>>>(qws, kws, vws, hws);
  k_proj<<<dim3(NN / 128, BB), 256, 0, stream>>>(hws, wph, p_b, x, (float*)d_out);
}
